// SimpleMambaModel_67448166417105
// MI455X (gfx1250) — compile-verified, matrix operand order checked by reading, not run on hardware
//
#include <hip/hip_runtime.h>
#include <hip/hip_bf16.h>

// ---------------------------------------------------------------------------
// Mamba2-ish model for MI455X (gfx1250, wave32, WMMA bf16 16x16x32,
// async global->LDS copies, global prefetch)
// ---------------------------------------------------------------------------

typedef __bf16 bf16_t;
typedef __attribute__((ext_vector_type(16))) __bf16 v16bf;
typedef __attribute__((ext_vector_type(8)))  __bf16 v8bf;
typedef __attribute__((ext_vector_type(8)))  float  v8f;

#define BSZ     512
#define NCH     129
#define NT      250
#define BT      (BSZ * NT)        // 128000 rows of (b,t)
#define DMODEL  128
#define DINNER  256
#define DSTATE  16
#define NHEADS  4
#define HEADDIM 64
#define CONVDIM 288               // DINNER + 2*DSTATE
#define DINPROJ 548               // 2*DINNER + 2*DSTATE + NHEADS
#define KMIX    160               // NCH=129 zero-padded to 5 chunks of 32
#define NPROJ   560               // DINPROJ padded to 35 tiles of 16
#define TILE_M  64

// ---- workspace layout (bytes, all 256B aligned) ----
#define WS_Z      ((size_t)0)                          // [BT][256] f32
#define WS_XBCR   ((size_t)131072000)                  // [BT][288] f32 (pre-conv)
#define WS_DT     ((size_t)278528000)                  // [BT][4]   f32 (softplus'd)
#define WS_XBCC   ((size_t)280576000)                  // [BT][288] f32 (post conv+silu)
#define WS_YB16   ((size_t)428032000)                  // [BT][256] bf16 (gated+normed)
#define WS_POOL   ((size_t)493568000)                  // [512][128] f32 accumulators
#define WS_WMIX   ((size_t)493830144)                  // [128][160] bf16
#define WS_WPROJ  ((size_t)493871104)                  // [560][128] bf16
#define WS_WOUT   ((size_t)494014464)                  // [128][256] bf16

// ---------------------------------------------------------------------------
// LDS byte offset of a shared-memory pointer (generic -> addrspace(3) cast
// strips the aperture; ptrtoint of an AS(3) pointer is the 32-bit LDS offset).
// ---------------------------------------------------------------------------
__device__ __forceinline__ unsigned lds_offset_of(const void* p) {
  return (unsigned)(unsigned long long)
      (const __attribute__((address_space(3))) char*)p;
}

// Async copy of 16 bytes global -> LDS (GLOBAL_LOAD_ASYNC_TO_LDS_B128,
// tracked by ASYNCcnt). vdst = LDS address VGPR, vaddr = 64-bit global addr.
__device__ __forceinline__ void async_copy_b128(void* lds_dst, const void* gsrc) {
  unsigned lds = lds_offset_of(lds_dst);
  unsigned long long ga = (unsigned long long)gsrc;
  asm volatile("global_load_async_to_lds_b128 %0, %1, off"
               :: "v"(lds), "v"(ga) : "memory");
}

__device__ __forceinline__ void async_wait_all() {
  asm volatile("s_wait_asynccnt 0x0" ::: "memory");
}

// ---------------------------------------------------------------------------
// WMMA bf16 fragment loader. CDNA5 16-bit A (16x32 MxK) per-lane layout:
//   M = lane&15, kbase = (lane&16)?8:0, elements 0..7 -> K=kbase..kbase+7,
//   elements 8..15 -> K=kbase+16..kbase+23  => two contiguous 16B loads.
// B (32x16 KxN) uses the mirrored mapping with N = lane&15, same K pattern.
// ---------------------------------------------------------------------------
__device__ __forceinline__ v16bf load_frag(const bf16_t* p) {
  v8bf lo = *(const v8bf*)(p);
  v8bf hi = *(const v8bf*)(p + 16);
  return __builtin_shufflevector(lo, hi, 0,1,2,3,4,5,6,7,8,9,10,11,12,13,14,15);
}

__device__ __forceinline__ v8f wmma_bf16(v16bf a, v16bf b, v8f c) {
  return __builtin_amdgcn_wmma_f32_16x16x32_bf16(
      /*neg_a=*/false, a, /*neg_b=*/false, b,
      /*c_mod=*/(short)0, c, /*reuse_a=*/false, /*reuse_b=*/false);
}

// ---------------------------------------------------------------------------
// K0: weight conversion to bf16 (zero-padded) + pooled-accumulator clear
// ---------------------------------------------------------------------------
__global__ void prep_kernel(const float* __restrict__ mixer_w,
                            const float* __restrict__ in_proj_w,
                            const float* __restrict__ out_proj_w,
                            bf16_t* __restrict__ wmix,
                            bf16_t* __restrict__ wproj,
                            bf16_t* __restrict__ wout,
                            float*  __restrict__ pooled) {
  const int N0 = DMODEL * KMIX;       // 20480
  const int N1 = NPROJ * DMODEL;      // 71680
  const int N2 = DMODEL * DINNER;     // 32768
  const int N3 = BSZ * DMODEL;        // 65536
  const int total = N0 + N1 + N2 + N3;
  for (int i = blockIdx.x * blockDim.x + threadIdx.x; i < total;
       i += gridDim.x * blockDim.x) {
    int j = i;
    if (j < N0) {
      int d = j / KMIX, c = j % KMIX;
      wmix[j] = (bf16_t)((c < NCH) ? mixer_w[d * NCH + c] : 0.f);
      continue;
    }
    j -= N0;
    if (j < N1) {
      int n = j / DMODEL;
      wproj[j] = (bf16_t)((n < DINPROJ) ? in_proj_w[j] : 0.f);
      continue;
    }
    j -= N1;
    if (j < N2) { wout[j] = (bf16_t)out_proj_w[j]; continue; }
    j -= N2;
    pooled[j] = 0.f;
  }
}

// ---------------------------------------------------------------------------
// K1: fused  u = mixer(x) + b  ->  zxbcdt = u @ in_proj_w.T  (WMMA bf16)
// One workgroup (8 waves) per 64-row (b,t) tile.
// ---------------------------------------------------------------------------
__global__ __launch_bounds__(256) void mixer_inproj_kernel(
    const float*  __restrict__ x,        // [512][129][250]
    const float*  __restrict__ mixer_b,  // [128]
    const bf16_t* __restrict__ wmix,     // [128][160]
    const bf16_t* __restrict__ wproj,    // [560][128]
    const float*  __restrict__ dt_bias,  // [4]
    float* __restrict__ z_out,           // [BT][256]
    float* __restrict__ xbc_out,         // [BT][288]
    float* __restrict__ dt_out) {        // [BT][4]
  __shared__ __align__(16) bf16_t Xt[TILE_M][KMIX + 8];   // X tile, bf16
  __shared__ __align__(16) bf16_t Ut[TILE_M][DMODEL + 8]; // u tile, bf16

  const int tid  = threadIdx.x;
  const int lane = tid & 31;
  const int wave = tid >> 5;
  const int row0 = blockIdx.x * TILE_M;
  const int mt  = wave & 3;     // M sub-tile (4 x 16 rows)
  const int grp = wave >> 2;    // 0/1: N-tile group

  // ---- stage X tile into LDS (coalesced over t) with K zero-padding ----
  for (int idx = tid; idx < TILE_M * KMIX; idx += 256) {
    int c = idx / TILE_M;           // 0..159
    int r = idx - c * TILE_M;       // 0..63
    int bt = row0 + r;
    int b = bt / NT, t = bt - b * NT;
    float v = 0.f;
    if (c < NCH) v = x[((size_t)b * NCH + c) * NT + t];
    Xt[r][c] = (bf16_t)v;
  }
  __syncthreads();

  const int kb    = (lane & 16) ? 8 : 0;   // per-lane K sub-offset
  const int mrow  = mt * 16 + (lane & 15); // A-fragment row
  const int nlane = lane & 15;             // B/C/D column within tile
  const int crow0 = mt * 16 + ((lane & 16) ? 8 : 0); // C/D row base

  // ---- stage 1: u[64][128] = X[64][160] x mixer_w^T ----
  v8f accU[4] = {v8f{}, v8f{}, v8f{}, v8f{}};
  for (int kc = 0; kc < KMIX; kc += 32) {
    v16bf a = load_frag(&Xt[mrow][kc + kb]);
#pragma unroll
    for (int j = 0; j < 4; ++j) {
      int n0 = (grp * 4 + j) * 16;
      v16bf bfr = load_frag(wmix + (size_t)(n0 + nlane) * KMIX + kc + kb);
      accU[j] = wmma_bf16(a, bfr, accU[j]);
    }
  }
#pragma unroll
  for (int j = 0; j < 4; ++j) {
    int n = (grp * 4 + j) * 16 + nlane;
    float bias = mixer_b[n];
#pragma unroll
    for (int v = 0; v < 8; ++v)
      Ut[crow0 + v][n] = (bf16_t)(accU[j][v] + bias);
  }
  __syncthreads();

  // ---- stage 2: zxbcdt[64][548] = u[64][128] x in_proj_w^T ----
  for (int nt = grp; nt < NPROJ / 16; nt += 2) {
    v8f acc = v8f{};
#pragma unroll
    for (int kc = 0; kc < DMODEL; kc += 32) {
      v16bf a   = load_frag(&Ut[mrow][kc + kb]);
      v16bf bfr = load_frag(wproj + (size_t)(nt * 16 + nlane) * DMODEL + kc + kb);
      acc = wmma_bf16(a, bfr, acc);
    }
    int ncol = nt * 16 + nlane;
#pragma unroll
    for (int v = 0; v < 8; ++v) {
      size_t bt = (size_t)(row0 + crow0 + v);
      float val = acc[v];
      if (ncol < DINNER) {
        z_out[bt * DINNER + ncol] = val;
      } else if (ncol < DINNER + CONVDIM) {
        xbc_out[bt * CONVDIM + (ncol - DINNER)] = val;
      } else if (ncol < DINPROJ) {
        float s = val + dt_bias[ncol - (DINNER + CONVDIM)];
        dt_out[bt * NHEADS + (ncol - (DINNER + CONVDIM))] =
            (s > 20.f) ? s : log1pf(expf(s));
      }
    }
  }
}

// ---------------------------------------------------------------------------
// K2: depthwise causal conv (D_CONV=4) + bias + SiLU, elementwise
// ---------------------------------------------------------------------------
__global__ void conv_silu_kernel(const float* __restrict__ xbc_raw,
                                 const float* __restrict__ conv_w,   // [288][4]
                                 const float* __restrict__ conv_b,   // [288]
                                 float* __restrict__ xbc_c) {
  size_t i = (size_t)blockIdx.x * blockDim.x + threadIdx.x;
  if (i >= (size_t)BT * CONVDIM) return;
  int j = (int)(i % CONVDIM);
  size_t bt = i / CONVDIM;
  int b = (int)(bt / NT), t = (int)(bt - (size_t)b * NT);
  // hint the deepest tap's row into cache ahead of the gathered reads
  __builtin_prefetch(&xbc_raw[((size_t)b * NT + (t > 3 ? t - 3 : 0)) * CONVDIM + j], 0, 0);
  float acc = conv_b[j];
#pragma unroll
  for (int k = 0; k < 4; ++k) {
    int tt = t + k - 3;
    if (tt >= 0)
      acc += conv_w[j * 4 + k] * xbc_raw[((size_t)b * NT + tt) * CONVDIM + j];
  }
  xbc_c[i] = acc / (1.f + expf(-acc));     // SiLU
}

// ---------------------------------------------------------------------------
// K3: sequential SSM scan over T, fused with gating (silu(z)) and RMSNorm.
// One workgroup per batch element; thread = (head, headdim) lane, 16-wide
// state kept in registers; B/C/dt staged through LDS each step.
// ---------------------------------------------------------------------------
__global__ __launch_bounds__(256) void scan_gate_norm_kernel(
    const float* __restrict__ xbc_c,   // [BT][288]
    const float* __restrict__ dt,      // [BT][4]
    const float* __restrict__ z,       // [BT][256]
    const float* __restrict__ A_log,   // [4]
    const float* __restrict__ Dp,      // [4]
    const float* __restrict__ norm_w,  // [256]
    bf16_t* __restrict__ yb16) {       // [BT][256]
  __shared__ float sB[DSTATE], sC[DSTATE], sdt[NHEADS], spart[8];
  const int b = blockIdx.x;
  const int tid = threadIdx.x;        // == h*64 + p
  const int h = tid >> 6;
  const float Ah = -expf(A_log[h]);
  const float Dh = Dp[h];
  const float nw = norm_w[tid];

  float s[DSTATE];
#pragma unroll
  for (int n = 0; n < DSTATE; ++n) s[n] = 0.f;

  for (int t = 0; t < NT; ++t) {
    const size_t bt = (size_t)b * NT + t;
    // prefetch next timestep's activations while this step computes
    if (t + 1 < NT) {
      __builtin_prefetch(&xbc_c[(bt + 1) * CONVDIM + tid], 0, 0);
      __builtin_prefetch(&z[(bt + 1) * DINNER + tid], 0, 0);
    }
    __syncthreads();                               // protect LDS reuse
    if (tid < 16)      sB[tid]      = xbc_c[bt * CONVDIM + DINNER + tid];
    else if (tid < 32) sC[tid - 16] = xbc_c[bt * CONVDIM + DINNER + DSTATE + (tid - 16)];
    else if (tid < 36) sdt[tid - 32] = dt[bt * NHEADS + (tid - 32)];
    __syncthreads();

    float x_t = xbc_c[bt * CONVDIM + tid];
    float dth = sdt[h];
    float dA  = expf(dth * Ah);
    float dx  = dth * x_t;
    float y = 0.f;
#pragma unroll
    for (int n = 0; n < DSTATE; ++n) {
      s[n] = s[n] * dA + dx * sB[n];
      y += s[n] * sC[n];
    }
    y += Dh * x_t;

    // gate with silu(z)
    float zv = z[bt * DINNER + tid];
    float g = y * (zv / (1.f + expf(-zv)));

    // RMS over the 256 channels of this (b,t) row
    float ss = g * g;
#pragma unroll
    for (int m = 16; m > 0; m >>= 1) ss += __shfl_xor(ss, m, 32);
    if ((tid & 31) == 0) spart[tid >> 5] = ss;
    __syncthreads();
    float tot = 0.f;
#pragma unroll
    for (int w = 0; w < 8; ++w) tot += spart[w];
    float r = rsqrtf(tot * (1.f / DINNER) + 1e-5f);
    yb16[bt * DINNER + tid] = (bf16_t)(g * r * nw);
  }
}

// ---------------------------------------------------------------------------
// K4: out_proj (256 -> 128) WMMA + fused mean pooling via f32 atomics.
// Y tile is staged global->LDS with GLOBAL_LOAD_ASYNC_TO_LDS_B128 (ASYNCcnt).
// ---------------------------------------------------------------------------
__global__ __launch_bounds__(256) void outproj_pool_kernel(
    const bf16_t* __restrict__ yb16,   // [BT][256]
    const bf16_t* __restrict__ wout,   // [128][256]
    float* __restrict__ pooled) {      // [512][128]
  __shared__ __align__(16) bf16_t Yt[TILE_M][DINNER + 8];
  const int tid  = threadIdx.x;
  const int lane = tid & 31;
  const int wave = tid >> 5;
  const int row0 = blockIdx.x * TILE_M;
  const int mt  = wave & 3;
  const int grp = wave >> 2;

  // async bulk copy: 64 rows x 256 bf16 = 32KB, 16B per lane per iteration
  for (int idx = tid; idx < TILE_M * (DINNER / 8); idx += 256) {
    int r = idx >> 5;
    int k8 = (idx & 31) * 8;
    async_copy_b128(&Yt[r][k8], yb16 + (size_t)(row0 + r) * DINNER + k8);
  }
  async_wait_all();
  __syncthreads();

  const int kb    = (lane & 16) ? 8 : 0;
  const int mrow  = mt * 16 + (lane & 15);
  const int nlane = lane & 15;
  const int crow0 = mt * 16 + ((lane & 16) ? 8 : 0);

  v8f acc[4] = {v8f{}, v8f{}, v8f{}, v8f{}};
  for (int kc = 0; kc < DINNER; kc += 32) {
    v16bf a = load_frag(&Yt[mrow][kc + kb]);
#pragma unroll
    for (int j = 0; j < 4; ++j) {
      int d0 = (grp * 4 + j) * 16;
      v16bf bfr = load_frag(wout + (size_t)(d0 + nlane) * DINNER + kc + kb);
      acc[j] = wmma_bf16(a, bfr, acc[j]);
    }
  }
#pragma unroll
  for (int j = 0; j < 4; ++j) {
    int d = (grp * 4 + j) * 16 + nlane;
#pragma unroll
    for (int v = 0; v < 8; ++v) {
      int bt = row0 + crow0 + v;
      int b = bt / NT;
      atomicAdd(&pooled[b * DMODEL + d], acc[j][v] * (1.f / NT));
    }
  }
}

// ---------------------------------------------------------------------------
// K5: head projection  out[b] = pooled[b] . head_w + head_b
// ---------------------------------------------------------------------------
__global__ __launch_bounds__(128) void head_kernel(
    const float* __restrict__ pooled, const float* __restrict__ head_w,
    const float* __restrict__ head_b, float* __restrict__ out) {
  __shared__ float part[4];
  const int b = blockIdx.x, tid = threadIdx.x;
  float v = pooled[b * DMODEL + tid] * head_w[tid];
#pragma unroll
  for (int m = 16; m > 0; m >>= 1) v += __shfl_xor(v, m, 32);
  if ((tid & 31) == 0) part[tid >> 5] = v;
  __syncthreads();
  if (tid == 0) out[b] = part[0] + part[1] + part[2] + part[3] + head_b[0];
}

// ---------------------------------------------------------------------------
extern "C" void kernel_launch(void* const* d_in, const int* in_sizes, int n_in,
                              void* d_out, int out_size, void* d_ws, size_t ws_size,
                              hipStream_t stream) {
  const float* x         = (const float*)d_in[0];
  const float* mixer_w   = (const float*)d_in[1];
  const float* mixer_b   = (const float*)d_in[2];
  const float* in_proj_w = (const float*)d_in[3];
  const float* conv_w    = (const float*)d_in[4];
  const float* conv_b    = (const float*)d_in[5];
  const float* dt_bias   = (const float*)d_in[6];
  const float* A_log     = (const float*)d_in[7];
  const float* Dp        = (const float*)d_in[8];
  const float* norm_w    = (const float*)d_in[9];
  const float* out_proj_w= (const float*)d_in[10];
  const float* head_w    = (const float*)d_in[11];
  const float* head_b    = (const float*)d_in[12];

  char* ws = (char*)d_ws;
  float*  z_buf   = (float*)(ws + WS_Z);
  float*  xbcr    = (float*)(ws + WS_XBCR);
  float*  dt_buf  = (float*)(ws + WS_DT);
  float*  xbcc    = (float*)(ws + WS_XBCC);
  bf16_t* yb16    = (bf16_t*)(ws + WS_YB16);
  float*  pooled  = (float*)(ws + WS_POOL);
  bf16_t* wmix    = (bf16_t*)(ws + WS_WMIX);
  bf16_t* wproj   = (bf16_t*)(ws + WS_WPROJ);
  bf16_t* wout    = (bf16_t*)(ws + WS_WOUT);

  prep_kernel<<<256, 256, 0, stream>>>(mixer_w, in_proj_w, out_proj_w,
                                       wmix, wproj, wout, pooled);

  mixer_inproj_kernel<<<BT / TILE_M, 256, 0, stream>>>(
      x, mixer_b, wmix, wproj, dt_bias, z_buf, xbcr, dt_buf);

  {
    size_t total = (size_t)BT * CONVDIM;
    int blocks = (int)((total + 255) / 256);
    conv_silu_kernel<<<blocks, 256, 0, stream>>>(xbcr, conv_w, conv_b, xbcc);
  }

  scan_gate_norm_kernel<<<BSZ, 256, 0, stream>>>(xbcc, dt_buf, z_buf, A_log,
                                                 Dp, norm_w, yb16);

  outproj_pool_kernel<<<BT / TILE_M, 256, 0, stream>>>(yb16, wout, pooled);

  head_kernel<<<BSZ, 128, 0, stream>>>(pooled, head_w, head_b, (float*)d_out);
}
